// NeuralODE_37280316129787
// MI455X (gfx1250) — matrix operand order repaired and checked by reading, not compiled, not run on hardware
//
#include <hip/hip_runtime.h>
#include <hip/hip_bf16.h>
#include <math.h>

// ---------------------------------------------------------------------------
// NeuralODE on gfx1250 (MI455X).
//  - Only x[:, S-1, :] matters (h0 = x_last @ Wi + bi): read 1 MB, not 209 MB.
//  - 64 DOPRI5 steps x 6 stages x 2 (128x128) GEMMs, fully on-chip per block.
//  - GEMMs via v_wmma_f32_16x16x32_f16; weights staged once into LDS as f16,
//    n-major with padded stride (136 halves) for conflict-free B fragments.
//  - Branch-free tanh (v_exp_f32 + v_rcp_f32), no EXEC divergence in the loop.
//  - Combine phase is wave-local over its own N-half -> 2 barriers/stage.
// ---------------------------------------------------------------------------

typedef __attribute__((ext_vector_type(16))) _Float16 v16h;
typedef __attribute__((ext_vector_type(8)))  _Float16 v8h;
typedef __attribute__((ext_vector_type(8)))  float    v8f;

#define NSTEPS   64
#define BTOT     4096
#define SEQ      200
#define DIN      64
#define H        128
#define LDH      136      // padded f16 row stride (halves): 272B -> bank stride 4

// LDS layout (bytes)
#define OFF_W1T  0                         // 128 x LDH f16  = 34816
#define OFF_W2T  34816                     // 128 x LDH f16  = 34816
#define OFF_B1   69632                     // 128 f32        = 512
#define OFF_B2   70144                     // 128 f32        = 512
#define OFF_KS   70656                     // 6 x 2048 f32   = 49152
#define OFF_Y0   119808                    // 2048 f32       = 8192
#define OFF_YB   128000                    // 16 x LDH f16   = 4352
#define OFF_TB   132352                    // 16 x LDH f16   = 4352
#define SMEM_BYTES 136704

// Dormand-Prince stage-combine coefficients: row s -> coeffs of k1..k_{s+1}
__constant__ float A_TAB[6][6] = {
  {1.0f/5.0f, 0.f, 0.f, 0.f, 0.f, 0.f},
  {3.0f/40.0f, 9.0f/40.0f, 0.f, 0.f, 0.f, 0.f},
  {44.0f/45.0f, -56.0f/15.0f, 32.0f/9.0f, 0.f, 0.f, 0.f},
  {19372.0f/6561.0f, -25360.0f/2187.0f, 64448.0f/6561.0f, -212.0f/729.0f, 0.f, 0.f},
  {9017.0f/3168.0f, -355.0f/33.0f, 46732.0f/5247.0f, 49.0f/176.0f, -5103.0f/18656.0f, 0.f},
  {35.0f/384.0f, 0.0f, 500.0f/1113.0f, 125.0f/192.0f, -2187.0f/6784.0f, 11.0f/84.0f},
};

// Branch-free tanh: tanh(x) = 1 - 2/(exp(2x)+1).
// v_exp_f32 + v_rcp_f32; saturates to +/-1 at +/-inf, no EXEC manipulation.
__device__ __forceinline__ float fast_tanh(float x) {
  float e = __builtin_amdgcn_exp2f(x * 2.885390081777926814f); // 2*log2(e)*x
  return 1.0f - 2.0f * __builtin_amdgcn_rcpf(e + 1.0f);
}

// One 16(M) x 64(N per wave) x 128(K) GEMM: A from f16 LDS tile (row stride
// LDH), B from f16 n-major LDS weights (row stride LDH). 16 v_wmma ops.
// A fragment (16-bit A 16x32): lane M holds K {kb*32+0..7, kb*32+16..23},
// lane M+16 holds K {kb*32+8..15, kb*32+24..31} -> two 16B ds loads per lane.
// B fragment (16-bit B 32x16): lane N holds K {kb*32+0..7, kb*32+16..23} like A (lanes 16-31
// hold K {kb*32+8..15, kb*32+24..31}) -> two contiguous 16B ds loads per lane.
__device__ __forceinline__ void gemm_16xN64xK128(
    const _Float16* Abuf, const _Float16* Bt, int lane, int nbase, v8f acc[4])
{
  const int arow = lane & 15;
  const int ahl  = (lane >> 4) << 3;   // 0 or 8  (K half-offset inside 16)
  const int bn   = lane & 15;
  const int bko  = (lane >> 4) << 3;   // 0 or 8 (K offset inside 32)
#pragma unroll
  for (int kb = 0; kb < 4; ++kb) {
    const _Float16* ap = Abuf + arow * LDH + kb * 32 + ahl;
    v8h alo = *(const v8h*)ap;
    v8h ahi = *(const v8h*)(ap + 16);
    v16h a = __builtin_shufflevector(alo, ahi,
              0,1,2,3,4,5,6,7,8,9,10,11,12,13,14,15);
#pragma unroll
    for (int t = 0; t < 4; ++t) {
      const _Float16* bp = Bt + (nbase + t * 16 + bn) * LDH + kb * 32 + bko;
      v8h blo = *(const v8h*)bp;
      v8h bhi = *(const v8h*)(bp + 16);
      v16h b = __builtin_shufflevector(blo, bhi,
                0,1,2,3,4,5,6,7,8,9,10,11,12,13,14,15);
      acc[t] = __builtin_amdgcn_wmma_f32_16x16x32_f16(
                 false, a, false, b, (short)0, acc[t], false, false);
    }
  }
}

__global__ __launch_bounds__(64)
void node_ode_kernel(const float* __restrict__ x,  const float* __restrict__ Wi,
                     const float* __restrict__ bi, const float* __restrict__ W1,
                     const float* __restrict__ b1g,const float* __restrict__ W2,
                     const float* __restrict__ b2g,const float* __restrict__ Wo,
                     const float* __restrict__ bo, float* __restrict__ out)
{
  extern __shared__ char smem[];
  _Float16* W1t  = (_Float16*)(smem + OFF_W1T);
  _Float16* W2t  = (_Float16*)(smem + OFF_W2T);
  float*    b1   = (float*)   (smem + OFF_B1);
  float*    b2   = (float*)   (smem + OFF_B2);
  float*    ks   = (float*)   (smem + OFF_KS);   // [6][16*H]
  float*    y0   = (float*)   (smem + OFF_Y0);   // [16*H]
  _Float16* ybuf = (_Float16*)(smem + OFF_YB);   // [16][LDH] stage input (f16)
  _Float16* tbuf = (_Float16*)(smem + OFF_TB);   // [16][LDH] tanh intermediate

  const int tid   = threadIdx.x;
  const int lane  = tid & 31;
  const int wave  = tid >> 5;
  const int nbase = wave * 64;       // wave 0 -> N 0..63, wave 1 -> N 64..127
  const int b0    = blockIdx.x * 16; // 16 batch rows per block
  const float dt  = (float)(SEQ - 1) / (float)NSTEPS;

  // --- one-time staging: transpose W1/W2 into LDS as f16 (n-major, padded) ---
  for (int i = tid; i < H * H; i += 64) {
    int k = i >> 7, n = i & 127;     // i = k*128 + n (coalesced global reads)
    W1t[n * LDH + k] = (_Float16)W1[i];
    W2t[n * LDH + k] = (_Float16)W2[i];
  }
  for (int i = tid; i < H; i += 64) { b1[i] = b1g[i]; b2[i] = b2g[i]; }

  // --- stage last time-slice of x, then h0 = x_last @ Wi + bi ---
  float* xs = ks;                    // reuse k-storage before the loop
  for (int i = tid; i < 16 * DIN; i += 64) {
    int m = i >> 6, d = i & 63;
    xs[i] = x[((size_t)(b0 + m) * SEQ + (SEQ - 1)) * DIN + d];
  }
  __syncthreads();
  for (int i = tid; i < 16 * H; i += 64) {
    int m = i >> 7, n = i & 127;
    float acc = bi[n];
    for (int d = 0; d < DIN; ++d) acc += xs[m * DIN + d] * Wi[d * H + n];
    y0[i] = acc;
    ybuf[m * LDH + n] = (_Float16)acc;
  }
  __syncthreads();

  // --- 64 DOPRI5 steps, entirely on-chip ---
  for (int step = 0; step < NSTEPS; ++step) {
#pragma unroll 1
    for (int s = 0; s < 6; ++s) {
      // GEMM1: tbuf = f16(tanh(ybuf @ W1 + b1))  (writes own N-half)
      {
        v8f acc[4] = {};
        gemm_16xN64xK128(ybuf, W1t, lane, nbase, acc);
#pragma unroll
        for (int t = 0; t < 4; ++t) {
          int n = nbase + t * 16 + (lane & 15);
          float bias = b1[n];
#pragma unroll
          for (int v = 0; v < 8; ++v) {
            int m = v + ((lane >> 4) << 3);
            tbuf[m * LDH + n] = (_Float16)fast_tanh(acc[t][v] + bias);
          }
        }
      }
      __syncthreads();   // tbuf ready (GEMM2 reads full K range)
      // GEMM2: k_s = tbuf @ W2 + b2 -> ks[s] (own N-half, f32)
      {
        v8f acc[4] = {};
        gemm_16xN64xK128(tbuf, W2t, lane, nbase, acc);
        float* kdst = ks + s * (16 * H);
#pragma unroll
        for (int t = 0; t < 4; ++t) {
          int n = nbase + t * 16 + (lane & 15);
          float bias = b2[n];
#pragma unroll
          for (int v = 0; v < 8; ++v) {
            int m = v + ((lane >> 4) << 3);
            kdst[m * H + n] = acc[t][v] + bias;
          }
        }
      }
      // Combine (wave-local: this wave only touches its own N-half, which it
      // itself wrote to ks[0..s], so no barrier needed after the ks stores).
      for (int i = lane; i < 16 * 64; i += 32) {
        int m = i >> 6, n = nbase + (i & 63);
        int idx = m * H + n;
        float sum = 0.f;
        for (int j = 0; j <= s; ++j) sum += A_TAB[s][j] * ks[j * (16 * H) + idx];
        float yv = y0[idx] + dt * sum;
        if (s == 5) y0[idx] = yv;
        ybuf[m * LDH + n] = (_Float16)yv;
      }
      __syncthreads();   // ybuf ready (next GEMM1 reads full K range);
                         // also orders tbuf reads vs next stage's overwrite
    }
  }

  // --- readout: out = h_final @ Wo + bo (16 rows per block) ---
  if (tid < 16) {
    float acc = bo[0];
    for (int hh = 0; hh < H; ++hh) acc += y0[tid * H + hh] * Wo[hh];
    out[b0 + tid] = acc;
  }
}

extern "C" void kernel_launch(void* const* d_in, const int* in_sizes, int n_in,
                              void* d_out, int out_size, void* d_ws, size_t ws_size,
                              hipStream_t stream) {
  (void)in_sizes; (void)n_in; (void)d_ws; (void)ws_size; (void)out_size;
  const float* x  = (const float*)d_in[0];
  const float* Wi = (const float*)d_in[1];
  const float* bi = (const float*)d_in[2];
  const float* W1 = (const float*)d_in[3];
  const float* b1 = (const float*)d_in[4];
  const float* W2 = (const float*)d_in[5];
  const float* b2 = (const float*)d_in[6];
  const float* Wo = (const float*)d_in[7];
  const float* bo = (const float*)d_in[8];
  float* out = (float*)d_out;

  (void)hipFuncSetAttribute((const void*)node_ode_kernel,
                            hipFuncAttributeMaxDynamicSharedMemorySize,
                            (int)SMEM_BYTES);
  node_ode_kernel<<<BTOT / 16, 64, SMEM_BYTES, stream>>>(
      x, Wi, bi, W1, b1, W2, b2, Wo, bo, out);
}
